// FlexAttention_5196910428389
// MI455X (gfx1250) — matrix operand order repaired and checked by reading, not compiled, not run on hardware
//
#include <hip/hip_runtime.h>

typedef __attribute__((ext_vector_type(8)))  float  v8f;
typedef __attribute__((ext_vector_type(16))) __bf16 v16bf;
typedef __attribute__((__vector_size__(4 * sizeof(int)))) int vi4;   // matches builtin param

#define B_    2
#define HQ_   16
#define HKV_  4
#define S_    2048
#define D_    128
#define W_    512
#define TQ    128      // queries per block
#define CK    32       // keys per chunk
#define VTS   40       // V^T LDS row stride (bf16 elements); mult of 8 -> 16B aligned rows

#if __has_builtin(__builtin_amdgcn_global_load_async_to_lds_b128)
#define HAVE_ASYNC 1
#else
#define HAVE_ASYNC 0
#endif

// AS-qualified int4-vector pointer casts for the async-to-LDS builtin
#define AS1_V4I(p) ((__attribute__((address_space(1))) vi4*)(__attribute__((address_space(1))) void*)(void*)(const void*)(p))
#define AS3_V4I(p) ((__attribute__((address_space(3))) vi4*)(__attribute__((address_space(3))) void*)(void*)(p))

__device__ __forceinline__ unsigned short f2bf(float f) {
  union { float f; unsigned u; } x; x.f = f;
  unsigned r = x.u + 0x7FFFu + ((x.u >> 16) & 1u);   // round-to-nearest-even bf16
  return (unsigned short)(r >> 16);
}

__device__ __forceinline__ void wait_async0() {
#if __has_builtin(__builtin_amdgcn_s_wait_asynccnt)
  __builtin_amdgcn_s_wait_asynccnt(0);
#else
  asm volatile("s_wait_asynccnt 0" ::: "memory");
#endif
}

union AOp {                  // one 16x32 bf16 WMMA operand (8 VGPRs)
  uint4          q2[2];
  unsigned       u[8];
  unsigned short h[16];
  v16bf          v;
};

__global__ __launch_bounds__(256, 1) void flexattn_sliding_sink(
    const float* __restrict__ qg_, const float* __restrict__ kg_,
    const float* __restrict__ vg_, const float* __restrict__ sinkw,
    float* __restrict__ out)
{
  __shared__ __attribute__((aligned(16))) unsigned short ldsK[CK * D_];   // bf16 K chunk [32][128]
  __shared__ __attribute__((aligned(16))) unsigned short ldsVT[D_ * VTS]; // bf16 V^T [128][40]
#if HAVE_ASYNC
  __shared__ __attribute__((aligned(16))) float rawK[2][CK * D_];         // f32 staging, double buffered
  __shared__ __attribute__((aligned(16))) float rawV[2][CK * D_];
#endif

  const int tid  = threadIdx.x;
  const int lane = tid & 31;
  const int wv   = tid >> 5;                 // wave 0..7
  const int x    = lane & 15;                // query column / matrix row selector
  const int rb   = (lane < 16) ? 0 : 8;      // row-half selector for A/C layouts

  int bid = blockIdx.x;
  const int qt  = bid & 15; bid >>= 4;       // S_/TQ = 16 query tiles
  const int hq  = bid & 15; bid >>= 4;
  const int b   = bid;
  const int hkv = hq >> 2;                   // GQA group of 4

  const int q0 = qt * TQ;
  const int qw = q0 + wv * 16;               // wave's first query row
  const int qg = qw + x;                     // this lane's query row
  const float scale = 0.08838834764831845f;  // 1/sqrt(128)

  const float* qbase = qg_ + (((size_t)b * HQ_  + hq ) * S_) * D_;
  const float* kbase = kg_ + (((size_t)b * HKV_ + hkv) * S_) * D_;
  const float* vbase = vg_ + (((size_t)b * HKV_ + hkv) * S_) * D_;

  // ---- Q^T B-operands (32 d-rows x 16 queries each), 4 ops cover D=128 ----
  v16bf qop[4];
  {
    const float* qrow = qbase + (size_t)qg * D_;
    #pragma unroll
    for (int ob = 0; ob < 4; ++ob) {
      const int d0 = ob * 32 + ((lane < 16) ? 0 : 8);
      AOp a;
      #pragma unroll
      for (int i4 = 0; i4 < 4; ++i4) {
        float4 f = *(const float4*)(qrow + d0 + 4 * i4 + ((i4 >= 2) ? 8 : 0));
        a.h[4 * i4 + 0] = f2bf(f.x * scale);
        a.h[4 * i4 + 1] = f2bf(f.y * scale);
        a.h[4 * i4 + 2] = f2bf(f.z * scale);
        a.h[4 * i4 + 3] = f2bf(f.w * scale);
      }
      qop[ob] = a.v;
    }
  }

  v8f acc[8] = {};                 // O^T accumulators: 8 d-blocks x (16d x 16q)
  float mrow = -1.0e30f, lrow = 0.0f;

  int kfirst = q0 - W_; if (kfirst < 0) kfirst = 0;
  const int c0 = kfirst >> 5;
  const int c1 = (q0 + TQ - 1) >> 5;

#if HAVE_ASYNC
  // prologue: kick off async DMA of the first chunk into raw buffer parity c0&1
  {
    const float* kg = kbase + ((size_t)(c0 << 5)) * D_;
    const float* vg = vbase + ((size_t)(c0 << 5)) * D_;
    float* rk = rawK[c0 & 1];
    float* rv = rawV[c0 & 1];
    #pragma unroll
    for (int i = 0; i < 4; ++i) {
      const int f4 = (tid + 256 * i) << 2;          // float offset, 16B granules
      __builtin_amdgcn_global_load_async_to_lds_b128(AS1_V4I(kg + f4), AS3_V4I(rk + f4), 0, 0);
      __builtin_amdgcn_global_load_async_to_lds_b128(AS1_V4I(vg + f4), AS3_V4I(rv + f4), 0, 0);
    }
  }
#endif

  for (int c = c0; c <= c1; ++c) {
    const int kv0 = c << 5;

#if HAVE_ASYNC
    wait_async0();        // this wave's DMA for chunk c done
    __syncthreads();      // all waves' DMA done; previous compute reads of bf16 done
    if (c + 1 <= c1) {    // start DMA of next chunk into the other raw buffer
      const float* kg = kbase + (size_t)(kv0 + CK) * D_;
      const float* vg = vbase + (size_t)(kv0 + CK) * D_;
      float* rk = rawK[(c + 1) & 1];
      float* rv = rawV[(c + 1) & 1];
      #pragma unroll
      for (int i = 0; i < 4; ++i) {
        const int f4 = (tid + 256 * i) << 2;
        __builtin_amdgcn_global_load_async_to_lds_b128(AS1_V4I(kg + f4), AS3_V4I(rk + f4), 0, 0);
        __builtin_amdgcn_global_load_async_to_lds_b128(AS1_V4I(vg + f4), AS3_V4I(rv + f4), 0, 0);
      }
    }
    // ---- convert raw f32 chunk -> bf16 K (row major) and V^T ----
    {
      const float* rk = rawK[c & 1];
      const float* rv = rawV[c & 1];
      #pragma unroll
      for (int i = 0; i < 4; ++i) {
        const int f   = tid + 256 * i;       // float4 index 0..1023
        const int row = f >> 5;              // key within chunk
        const int c4  = (f & 31) << 2;       // d start
        float4 kf = *(const float4*)(rk + row * D_ + c4);
        unsigned p0 = (unsigned)f2bf(kf.x) | ((unsigned)f2bf(kf.y) << 16);
        unsigned p1 = (unsigned)f2bf(kf.z) | ((unsigned)f2bf(kf.w) << 16);
        *(uint2*)(&ldsK[row * D_ + c4]) = make_uint2(p0, p1);
        float4 vf = *(const float4*)(rv + row * D_ + c4);
        ldsVT[(c4 + 0) * VTS + row] = f2bf(vf.x);
        ldsVT[(c4 + 1) * VTS + row] = f2bf(vf.y);
        ldsVT[(c4 + 2) * VTS + row] = f2bf(vf.z);
        ldsVT[(c4 + 3) * VTS + row] = f2bf(vf.w);
      }
    }
    __syncthreads();
#else
    __syncthreads();
    {
      const float* kg = kbase + (size_t)kv0 * D_;
      const float* vg = vbase + (size_t)kv0 * D_;
      #pragma unroll
      for (int i = 0; i < 4; ++i) {
        const int f   = tid + 256 * i;
        const int row = f >> 5;
        const int c4  = (f & 31) << 2;
        float4 kf = *(const float4*)(kg + row * D_ + c4);
        unsigned p0 = (unsigned)f2bf(kf.x) | ((unsigned)f2bf(kf.y) << 16);
        unsigned p1 = (unsigned)f2bf(kf.z) | ((unsigned)f2bf(kf.w) << 16);
        *(uint2*)(&ldsK[row * D_ + c4]) = make_uint2(p0, p1);
        float4 vf = *(const float4*)(vg + row * D_ + c4);
        ldsVT[(c4 + 0) * VTS + row] = f2bf(vf.x);
        ldsVT[(c4 + 1) * VTS + row] = f2bf(vf.y);
        ldsVT[(c4 + 2) * VTS + row] = f2bf(vf.z);
        ldsVT[(c4 + 3) * VTS + row] = f2bf(vf.w);
      }
      if (c < c1) __builtin_prefetch(kbase + (size_t)(kv0 + CK) * D_, 0, 1);
    }
    __syncthreads();
#endif

    // wave-uniform skip: chunk outside this wave's band [qw-W, qw+15]
    if (kv0 > qw + 15 || kv0 + 31 < qw - W_) continue;

    // ---- S^T = K * Q^T : two 16-key subtiles, 2 independent partial chains each ----
    v8f p0a = {}, p0b = {}, p1a = {}, p1b = {};
    #pragma unroll
    for (int ob = 0; ob < 4; ++ob) {
      const int dsel = ob * 32 + rb;
      AOp a0, a1;
      a0.q2[0] = *(const uint4*)(&ldsK[( 0 + x) * D_ + dsel]);
      a0.q2[1] = *(const uint4*)(&ldsK[( 0 + x) * D_ + dsel + 16]);
      a1.q2[0] = *(const uint4*)(&ldsK[(16 + x) * D_ + dsel]);
      a1.q2[1] = *(const uint4*)(&ldsK[(16 + x) * D_ + dsel + 16]);
      if ((ob & 1) == 0) {
        p0a = __builtin_amdgcn_wmma_f32_16x16x32_bf16(false, a0.v, false, qop[ob],
                                                      (short)0, p0a, false, false);
        p1a = __builtin_amdgcn_wmma_f32_16x16x32_bf16(false, a1.v, false, qop[ob],
                                                      (short)0, p1a, false, false);
      } else {
        p0b = __builtin_amdgcn_wmma_f32_16x16x32_bf16(false, a0.v, false, qop[ob],
                                                      (short)0, p0b, false, false);
        p1b = __builtin_amdgcn_wmma_f32_16x16x32_bf16(false, a1.v, false, qop[ob],
                                                      (short)0, p1b, false, false);
      }
    }
    v8f sc0 = p0a + p0b;
    v8f sc1 = p1a + p1b;

    // ---- band mask + online softmax (lane-local rows; pair-combine by shfl 16) ----
    float s0[8], s1[8];
    float lm = -1.0e30f;
    #pragma unroll
    for (int r = 0; r < 8; ++r) {
      const int j0 = kv0 +      rb + r;
      const int j1 = kv0 + 16 + rb + r;
      s0[r] = ((j0 <= qg) && (qg - j0 <= W_)) ? sc0[r] : -1.0e30f;
      s1[r] = ((j1 <= qg) && (qg - j1 <= W_)) ? sc1[r] : -1.0e30f;
      lm = fmaxf(lm, fmaxf(s0[r], s1[r]));
    }
    lm = fmaxf(lm, __shfl_xor(lm, 16, 32));
    const float mnew = fmaxf(mrow, lm);
    const float fac  = __expf(mrow - mnew);
    float e0[8], e1[8], ls = 0.0f;
    #pragma unroll
    for (int r = 0; r < 8; ++r) {
      float t0 = __expf(s0[r] - mnew);
      float t1 = __expf(s1[r] - mnew);
      e0[r] = (s0[r] < -0.9e30f) ? 0.0f : t0;   // fully-masked stays 0 even if mnew==-1e30
      e1[r] = (s1[r] < -0.9e30f) ? 0.0f : t1;
      ls += e0[r] + e1[r];
    }
    ls += __shfl_xor(ls, 16, 32);
    lrow = lrow * fac + ls;
    mrow = mnew;
    #pragma unroll
    for (int t = 0; t < 8; ++t) acc[t] *= fac;

    // ---- build P^T B-operand (32 keys x 16 queries) from e0/e1 ----
    AOp pb;
    #pragma unroll
    for (int r = 0; r < 8; ++r) {
      const float o0 = __shfl_xor(e0[r], 16, 32);
      const float o1 = __shfl_xor(e1[r], 16, 32);
      const float lo8 = e0[r];   // K = 0..7  | 8..15
      const float hi8 = e1[r];   // K = 16..23 | 24..31
      pb.h[r]     = f2bf(lo8);
      pb.h[8 + r] = f2bf(hi8);
    }

    // ---- O^T += V^T * P^T : 8 independent d-block chains, K-dim = 32 keys ----
    #pragma unroll
    for (int t = 0; t < 8; ++t) {
      AOp av;
      const int row = 16 * t + x;                  // d row
      av.q2[0] = *(const uint4*)(&ldsVT[row * VTS + rb]);
      av.q2[1] = *(const uint4*)(&ldsVT[row * VTS + rb + 16]);
      acc[t] = __builtin_amdgcn_wmma_f32_16x16x32_bf16(false, av.v, false, pb.v,
                                                       (short)0, acc[t], false, false);
    }
  }

  // ---- sink joins the denominator only; normalize and store ----
  const float denom = lrow + __expf(sinkw[hq] - mrow);
  const float inv = 1.0f / denom;
  float* orow = out + (((size_t)b * HQ_ + hq) * S_ + qg) * D_;
  #pragma unroll
  for (int t = 0; t < 8; ++t) {
    float4 oA, oB;
    oA.x = acc[t][0] * inv; oA.y = acc[t][1] * inv;
    oA.z = acc[t][2] * inv; oA.w = acc[t][3] * inv;
    oB.x = acc[t][4] * inv; oB.y = acc[t][5] * inv;
    oB.z = acc[t][6] * inv; oB.w = acc[t][7] * inv;
    *(float4*)(orow + 16 * t + rb + 0) = oA;
    *(float4*)(orow + 16 * t + rb + 4) = oB;
  }
}

extern "C" void kernel_launch(void* const* d_in, const int* in_sizes, int n_in,
                              void* d_out, int out_size, void* d_ws, size_t ws_size,
                              hipStream_t stream) {
  const float* q    = (const float*)d_in[0];
  const float* k    = (const float*)d_in[1];
  const float* v    = (const float*)d_in[2];
  const float* sink = (const float*)d_in[3];
  float* out        = (float*)d_out;
  (void)in_sizes; (void)n_in; (void)d_ws; (void)ws_size; (void)out_size;

  const int nblocks = B_ * HQ_ * (S_ / TQ);  // 2*16*16 = 512
  flexattn_sliding_sink<<<nblocks, 256, 0, stream>>>(q, k, v, sink, out);
}
